// NestedGCN_31636729102752
// MI455X (gfx1250) — matrix operand order repaired and checked by reading, not compiled, not run on hardware
//
#include <hip/hip_runtime.h>
#include <hip/hip_bf16.h>

// ---- problem constants (match reference) ----
#define G_    10000
#define NI_   32
#define EI_   128
#define EO_   160000
#define FIN_  64
#define H_    128
#define FG_   128
#define NC_   32

typedef __bf16 bf16_t;
typedef bf16_t v16bf __attribute__((ext_vector_type(16)));
typedef float  v8f   __attribute__((ext_vector_type(8)));

// =====================================================================
// WMMA fragment helpers (wave32). A/B use contiguous K (not ISA table order); C/D per CDNA5 ISA 7.12.2.
// A: 16-bit 16x32 (MxK). lane&15 = M, hi = lane>>4:
//    j -> K = hi*16 + j (contiguous, same as B)
// B: 16-bit 32x16 (KxN). lane&15 = N, hi selects K 0-15 / 16-31.
// C/D: f32 16x16. lane&15 = N, VGPR r -> M = hi*8 + r.
// All loaders are branch-free: rowscale/bias are always-valid pointers,
// ReLU is a template parameter.
// =====================================================================

__device__ __forceinline__ v16bf load_a_frag_f32(const float* A, int lda, int m0, int k0,
                                                 const float* __restrict__ rowscale) {
  const int lane = threadIdx.x & 31;
  const int m = lane & 15, hi = lane >> 4;
  const float s = rowscale[m0 + m];
  const float* row = A + (size_t)(m0 + m) * lda + k0;
  v16bf a;
#pragma unroll
  for (int j = 0; j < 16; ++j) {
    const int k = hi * 16 + j;   // contiguous K order, matches B
    a[j] = (bf16_t)(row[k] * s);
  }
  return a;
}

__device__ __forceinline__ v16bf load_b_frag_f32(const float* B, int ldb, int k0, int n0) {
  const int lane = threadIdx.x & 31;
  const int n = lane & 15, hi = lane >> 4;
  const float* col = B + (size_t)(k0 + hi * 16) * ldb + n0 + n;
  v16bf b;
#pragma unroll
  for (int j = 0; j < 16; ++j) b[j] = (bf16_t)col[(size_t)j * ldb];
  return b;
}

template <bool RELU>
__device__ __forceinline__ void store_d_frag(float* D, int ldd, int m0, int n0,
                                             const v8f c, const float* __restrict__ bias) {
  const int lane = threadIdx.x & 31;
  const int n = lane & 15, hi = lane >> 4;
  const float bv = bias[n0 + n];
#pragma unroll
  for (int r = 0; r < 8; ++r) {
    float v = c[r] + bv;
    if (RELU) v = fmaxf(v, 0.0f);
    D[(size_t)(m0 + hi * 8 + r) * ldd + n0 + n] = v;
  }
}

// =====================================================================
// One inner GCN layer, fully specialized. agg/H live in LDS (stride H_).
//   agg[d] += h[s]*norm[s]  (ds_add_f32)
//   h = relu( diag(norm)*agg @ W + b )   via bf16 WMMA, f32 accumulate.
// =====================================================================
template <int Fi, int Fo, bool RELU>
__device__ __forceinline__ void inner_layer(float* __restrict__ sh_h,
                                            float* __restrict__ sh_agg,
                                            const float* __restrict__ sh_norm,
                                            const int* __restrict__ src,
                                            const int* __restrict__ dst,
                                            const float* __restrict__ W,
                                            const float* __restrict__ bias,
                                            int tid, int wave) {
  // zero agg (Fi is a compile-time constant -> shifts, no division)
#pragma unroll
  for (int i = tid; i < NI_ * Fi; i += 128)
    sh_agg[(i / Fi) * H_ + (i % Fi)] = 0.0f;
  __syncthreads();

  // edge scatter with LDS float atomics
  for (int i = tid; i < EI_ * Fi; i += 128) {
    const int e = i / Fi, f = i % Fi;
    const int s = src[e], d = dst[e];
    atomicAdd(&sh_agg[d * H_ + f], sh_h[s * H_ + f] * sh_norm[s]);
  }
  __syncthreads();

  // GEMM: 2 M-tiles x (Fo/16) N-tiles; waves stripe over N-tiles and
  // reuse the full-K set of B fragments across both M-tiles.
  constexpr int NT = Fo / 16;
  constexpr int KT = Fi / 32;
  for (int nt = wave; nt < NT; nt += 4) {
    v16bf bf[KT];
#pragma unroll
    for (int kk = 0; kk < KT; ++kk)
      bf[kk] = load_b_frag_f32(W, Fo, kk * 32, nt * 16);
#pragma unroll
    for (int mt = 0; mt < 2; ++mt) {
      v8f acc = {};
#pragma unroll
      for (int kk = 0; kk < KT; ++kk) {
        const v16bf a = load_a_frag_f32(sh_agg, H_, mt * 16, kk * 32, sh_norm);
        acc = __builtin_amdgcn_wmma_f32_16x16x32_bf16(false, a, false, bf[kk],
                                                      (short)0, acc, false, false);
      }
      store_d_frag<RELU>(sh_h, H_, mt * 16, nt * 16, acc, bias);
    }
  }
  __syncthreads();
}

// =====================================================================
// Inner stage: one block (4 wave32) per inner graph; graph lives in LDS.
// =====================================================================
__global__ __launch_bounds__(128)
void inner_gcn_kernel(const float* __restrict__ feat,   // [G,NI,FIN]
                      const int*   __restrict__ srcs,   // [G,EI]
                      const int*   __restrict__ dsts,   // [G,EI]
                      const float* __restrict__ W0, const float* __restrict__ b0,
                      const float* __restrict__ W1, const float* __restrict__ b1,
                      const float* __restrict__ W2, const float* __restrict__ b2,
                      float* __restrict__ out)          // [G,FG]
{
  __shared__ float sh_h[NI_ * H_];     // activations (stride H_)
  __shared__ float sh_agg[NI_ * H_];   // aggregated messages
  __shared__ float sh_norm[NI_];
  __shared__ float sh_deg[NI_];

  const int g    = blockIdx.x;
  const int tid  = threadIdx.x;
  const int wave = tid >> 5;

  const int*   src  = srcs + (size_t)g * EI_;
  const int*   dst  = dsts + (size_t)g * EI_;
  const float* f_in = feat + (size_t)g * NI_ * FIN_;

  // load features into LDS (stride H_ so all layers share one layout)
#pragma unroll
  for (int i = tid; i < NI_ * FIN_; i += 128)
    sh_h[(i / FIN_) * H_ + (i % FIN_)] = f_in[i];
  if (tid < NI_) sh_deg[tid] = 0.0f;
  __syncthreads();

  // degree from dst, norm = deg^-1/2 (deg==0 -> 1)
  atomicAdd(&sh_deg[dst[tid]], 1.0f);           // EI_ == blockDim.x == 128
  __syncthreads();
  if (tid < NI_) {
    const float d = sh_deg[tid];
    sh_norm[tid] = rsqrtf(d > 0.0f ? d : 1.0f);
  }
  __syncthreads();

  inner_layer<FIN_, H_, true >(sh_h, sh_agg, sh_norm, src, dst, W0, b0, tid, wave);
  inner_layer<H_,   H_, true >(sh_h, sh_agg, sh_norm, src, dst, W1, b1, tid, wave);
  inner_layer<H_,   FG_, false>(sh_h, sh_agg, sh_norm, src, dst, W2, b2, tid, wave);

  // graph readout: mean over the 32 nodes
  for (int f = tid; f < FG_; f += 128) {
    float s = 0.0f;
#pragma unroll
    for (int n = 0; n < NI_; ++n) s += sh_h[n * H_ + f];
    out[(size_t)g * FG_ + f] = s * (1.0f / NI_);
  }
}

// =====================================================================
// Outer stage helper kernels
// =====================================================================
__global__ void zero_f32(float* __restrict__ p, int n) {
  const int i = blockIdx.x * blockDim.x + threadIdx.x;
  if (i < n) p[i] = 0.0f;
}

__global__ void outer_deg_kernel(const int* __restrict__ dst, float* __restrict__ deg, int ne) {
  const int i = blockIdx.x * blockDim.x + threadIdx.x;
  if (i < ne) atomicAdd(&deg[dst[i]], 1.0f);
}

__global__ void deg_to_norm(const float* __restrict__ deg, float* __restrict__ nrm, int n) {
  const int i = blockIdx.x * blockDim.x + threadIdx.x;
  if (i < n) {
    const float d = deg[i];
    nrm[i] = rsqrtf(d > 0.0f ? d : 1.0f);
  }
}

// agg[dst] += h[src] * norm[src], float4 chunks per thread (global f32 atomics, L2-resident)
__global__ void outer_scatter(const float* __restrict__ h, const int* __restrict__ src,
                              const int* __restrict__ dst, const float* __restrict__ nrm,
                              float* __restrict__ agg, int ne) {
  const int i = blockIdx.x * blockDim.x + threadIdx.x;
  constexpr int chunks = H_ / 4;
  if (i >= ne * chunks) return;
  const int e = i / chunks;
  const int c = (i - e * chunks) << 2;
  const int s = src[e], d = dst[e];
  const float ns = nrm[s];
  const float4 hv = *(const float4*)(h + (size_t)s * H_ + c);
  float* ap = agg + (size_t)d * H_ + c;
  atomicAdd(ap + 0, hv.x * ns);
  atomicAdd(ap + 1, hv.y * ns);
  atomicAdd(ap + 2, hv.z * ns);
  atomicAdd(ap + 3, hv.w * ns);
}

// Out[m,:] = relu( (norm[m]*A[m,:]) @ W + b ).
// One block owns MT=5 M-tiles (80 rows; 10000 = 125*80 exact); waves
// stripe over N-tiles and reuse B fragments across all 5 M-tiles.
template <int Fi, int Fo, bool RELU, int MT>
__global__ __launch_bounds__(128)
void outer_gemm(const float* __restrict__ A,    // [G, Fi]
                const float* __restrict__ nrm,  // [G]
                const float* __restrict__ W,    // [Fi, Fo]
                const float* __restrict__ bias, // [Fo]
                float* __restrict__ Out)        // [G, Fo]
{
  const int wave = threadIdx.x >> 5;
  const int m0 = blockIdx.x * (MT * 16);
  constexpr int NT = Fo / 16;
  constexpr int KT = Fi / 32;
  for (int nt = wave; nt < NT; nt += 4) {
    v16bf bf[KT];
#pragma unroll
    for (int kk = 0; kk < KT; ++kk)
      bf[kk] = load_b_frag_f32(W, Fo, kk * 32, nt * 16);
#pragma unroll
    for (int mt = 0; mt < MT; ++mt) {
      v8f acc = {};
#pragma unroll
      for (int kk = 0; kk < KT; ++kk) {
        const v16bf a = load_a_frag_f32(A, Fi, m0 + mt * 16, kk * 32, nrm);
        acc = __builtin_amdgcn_wmma_f32_16x16x32_bf16(false, a, false, bf[kk],
                                                      (short)0, acc, false, false);
      }
      store_d_frag<RELU>(Out, Fo, m0 + mt * 16, nt * 16, acc, bias);
    }
  }
}

// =====================================================================
// launch
// =====================================================================
extern "C" void kernel_launch(void* const* d_in, const int* in_sizes, int n_in,
                              void* d_out, int out_size, void* d_ws, size_t ws_size,
                              hipStream_t stream) {
  const float* feat = (const float*)d_in[0];
  const int*   isrc = (const int*)d_in[1];
  const int*   idst = (const int*)d_in[2];
  const int*   osrc = (const int*)d_in[3];
  const int*   odst = (const int*)d_in[4];
  const float* Wi0 = (const float*)d_in[5];  const float* bi0 = (const float*)d_in[6];
  const float* Wi1 = (const float*)d_in[7];  const float* bi1 = (const float*)d_in[8];
  const float* Wi2 = (const float*)d_in[9];  const float* bi2 = (const float*)d_in[10];
  const float* Wo0 = (const float*)d_in[11]; const float* bo0 = (const float*)d_in[12];
  const float* Wo1 = (const float*)d_in[13]; const float* bo1 = (const float*)d_in[14];
  const float* Wo2 = (const float*)d_in[15]; const float* bo2 = (const float*)d_in[16];
  float* out = (float*)d_out;

  // workspace layout (floats): h0[G*H] | h1[G*H] | agg[G*H] | deg[G] | nrm[G]
  float* ws  = (float*)d_ws;
  float* h0  = ws;
  float* h1  = h0 + (size_t)G_ * H_;
  float* agg = h1 + (size_t)G_ * H_;
  float* deg = agg + (size_t)G_ * H_;
  float* nrm = deg + G_;

  // ---- inner GCN stack + per-graph mean readout -> h0 [G, 128] ----
  inner_gcn_kernel<<<G_, 128, 0, stream>>>(feat, isrc, idst,
                                           Wi0, bi0, Wi1, bi1, Wi2, bi2, h0);

  // ---- outer graph degree / norm ----
  zero_f32<<<(G_ + 255) / 256, 256, 0, stream>>>(deg, G_);
  outer_deg_kernel<<<(EO_ + 255) / 256, 256, 0, stream>>>(odst, deg, EO_);
  deg_to_norm<<<(G_ + 255) / 256, 256, 0, stream>>>(deg, nrm, G_);

  const int scat_threads = EO_ * (H_ / 4);
  const int gemm_blocks  = G_ / 80;             // MT=5 -> 125 blocks exactly

  // ---- outer layer 0: h0 -> h1 (128 -> 128, relu) ----
  zero_f32<<<(G_ * H_ + 255) / 256, 256, 0, stream>>>(agg, G_ * H_);
  outer_scatter<<<(scat_threads + 255) / 256, 256, 0, stream>>>(h0, osrc, odst, nrm, agg, EO_);
  outer_gemm<H_, H_, true, 5><<<gemm_blocks, 128, 0, stream>>>(agg, nrm, Wo0, bo0, h1);

  // ---- outer layer 1: h1 -> h0 (128 -> 128, relu) ----
  zero_f32<<<(G_ * H_ + 255) / 256, 256, 0, stream>>>(agg, G_ * H_);
  outer_scatter<<<(scat_threads + 255) / 256, 256, 0, stream>>>(h1, osrc, odst, nrm, agg, EO_);
  outer_gemm<H_, H_, true, 5><<<gemm_blocks, 128, 0, stream>>>(agg, nrm, Wo1, bo1, h0);

  // ---- outer layer 2: h0 -> out (128 -> 32, no relu) ----
  zero_f32<<<(G_ * H_ + 255) / 256, 256, 0, stream>>>(agg, G_ * H_);
  outer_scatter<<<(scat_threads + 255) / 256, 256, 0, stream>>>(h0, osrc, odst, nrm, agg, EO_);
  outer_gemm<H_, NC_, false, 5><<<gemm_blocks, 128, 0, stream>>>(agg, nrm, Wo2, bo2, out);
}
